// GAT_22531398435166
// MI455X (gfx1250) — matrix operand order repaired and checked by reading, not compiled, not run on hardware
//
#include <hip/hip_runtime.h>
#include <hip/hip_bf16.h>

// ---------- types ----------
typedef __bf16 bf16;
typedef __attribute__((ext_vector_type(16))) __bf16 v16bf;
typedef __attribute__((ext_vector_type(8)))  __bf16 v8bf;
typedef __attribute__((ext_vector_type(8)))  float  v8f;
typedef __attribute__((ext_vector_type(4)))  float  vf4;

// async global->LDS (gfx1250, ASYNCcnt) if the toolchain declares it.
// Probe-derived signature: (v4i addrspace(1)*, v4i addrspace(3)*, imm offset, imm cpol)
#if defined(__has_builtin)
#if __has_builtin(__builtin_amdgcn_global_load_async_to_lds_b128)
#define HAVE_ASYNC_LDS 1
#endif
#endif
#ifdef HAVE_ASYNC_LDS
typedef __attribute__((__vector_size__(4 * sizeof(int)))) int vsi4;
typedef __attribute__((address_space(1))) vsi4 g_vsi4;
typedef __attribute__((address_space(3))) vsi4 l_vsi4;
#define ASYNC_B128(gp, lp) \
  __builtin_amdgcn_global_load_async_to_lds_b128((g_vsi4*)(gp), (l_vsi4*)(lp), 0, 0)
#endif

// ---------- helpers ----------
__device__ __forceinline__ bf16 f2bf(float f) { return (bf16)f; }  // v_cvt_pk_bf16_f32

__device__ __forceinline__ v8f zero8() {
  v8f z;
#pragma unroll
  for (int i = 0; i < 8; i++) z[i] = 0.f;
  return z;
}

__device__ __forceinline__ v16bf zerobf() {
  v16bf z;
#pragma unroll
  for (int i = 0; i < 16; i++) z[i] = (bf16)0.f;
  return z;
}

__device__ __forceinline__ v8f wmma_bf16(v16bf a, v16bf b, v8f c) {
  // D = A(16x32 bf16) x B(32x16 bf16) + C(16x16 f32)
  return __builtin_amdgcn_wmma_f32_16x16x32_bf16(
      /*neg_a=*/false, a, /*neg_b=*/false, b,
      /*c_mod=*/(short)0, c, /*reuse_a=*/false, /*reuse_b=*/false);
}

// A-fragment (16x32, 16-bit): lane&15 = row M, lane>>4 selects K-base:
// elems [0..15] = K 2*koff+0..15 (koff = 8*(lane>>4))
__device__ __forceinline__ v16bf loadA_row(const float* __restrict__ p, int kc, int koff) {
  const float* q = p + kc * 32 + 2 * koff;
  vf4 a0 = *(const vf4*)(q);
  vf4 a1 = *(const vf4*)(q + 4);
  vf4 b0 = *(const vf4*)(q + 8);
  vf4 b1 = *(const vf4*)(q + 12);
  v16bf r;
#pragma unroll
  for (int i = 0; i < 4; i++) {
    r[i]      = f2bf(a0[i]);
    r[4 + i]  = f2bf(a1[i]);
    r[8 + i]  = f2bf(b0[i]);
    r[12 + i] = f2bf(b1[i]);
  }
  return r;
}

__device__ __forceinline__ float sigm(float x) { return 1.f / (1.f + __expf(-x)); }

// =====================================================================
// Kernel 1: SAGEConv LSTM aggregator + fc_self/fc_neigh + relu
// one wave = 16-node tile; 512 WMMAs (LSTM) + 8 WMMAs (fc) per tile
// =====================================================================
__global__ void __launch_bounds__(128) lstm_sage_kernel(
    const float* __restrict__ nfeat, const int* __restrict__ neigh,
    const float* __restrict__ Wih,  const float* __restrict__ Whh,
    const float* __restrict__ bih,  const float* __restrict__ bhh,
    const float* __restrict__ Wself, const float* __restrict__ bself,
    const float* __restrict__ Wneigh, float* __restrict__ hout)
{
  extern __shared__ __attribute__((aligned(64))) char smem[];
  bf16* sW = (bf16*)smem;                 // W_ih bf16 [256][64] then W_hh: 64KB
  bf16* sH = (bf16*)(smem + 65536);       // per-wave h staging [16][64] bf16
  const int tid = threadIdx.x, lane = tid & 31, w = tid >> 5;

  // stage weights (f32 -> bf16) into LDS, row-major [j][k]
  for (int i = tid * 4; i < 16384; i += 128 * 4) {
    vf4 a = *(const vf4*)(Wih + i);
    vf4 b = *(const vf4*)(Whh + i);
#pragma unroll
    for (int q = 0; q < 4; q++) {
      sW[i + q]         = f2bf(a[q]);
      sW[16384 + i + q] = f2bf(b[q]);
    }
  }
  __syncthreads();

  const int tile = blockIdx.x * 4 + w;
  if (tile >= 1250) return;

  const int row = lane & 15, hi = lane >> 4, koff = hi * 8;
  bf16* myH = sH + w * 1024;

  // gate biases: per lane, one per column tile jt (j = 16*jt + row)
  float breg[16];
#pragma unroll
  for (int jt = 0; jt < 16; jt++) {
    int j = jt * 16 + row;
    breg[jt] = bih[j] + bhh[j];
  }

  v8f cfrag[4];
#pragma unroll
  for (int t = 0; t < 4; t++) cfrag[t] = zero8();
  v16bf hA[2];
  hA[0] = zerobf(); hA[1] = zerobf();

  const int nodeBase = tile * 16;
  const int myNode = nodeBase + row;

  for (int t = 0; t < 8; t++) {
    // gather neighbor-t features as A fragments (two K-chunks of 32)
    int idx = neigh[myNode * 8 + t];
    const float* xp = nfeat + idx * 64;
    v16bf xA[2];
    xA[0] = loadA_row(xp, 0, koff);
    xA[1] = loadA_row(xp, 1, koff);

#pragma unroll
    for (int ht = 0; ht < 4; ht++) {     // hidden-unit tile (16 units)
      v8f g[4];
#pragma unroll
      for (int gi = 0; gi < 4; gi++) {   // i, f, g, o gates
        int jt = gi * 4 + ht;
        // B[k][j] = W[j][k]; B-frag: 16 contiguous k at (kc*32 + 16*hi)
        const bf16* bi0 = sW + (jt * 16 + row) * 64 + hi * 16;
        const bf16* bh0 = bi0 + 16384;
        v8f acc = zero8();
        acc = wmma_bf16(xA[0], *(const v16bf*)(bi0),      acc);
        acc = wmma_bf16(xA[1], *(const v16bf*)(bi0 + 32), acc);
        acc = wmma_bf16(hA[0], *(const v16bf*)(bh0),      acc);
        acc = wmma_bf16(hA[1], *(const v16bf*)(bh0 + 32), acc);
        g[gi] = acc;
      }
#pragma unroll
      for (int r = 0; r < 8; r++) {      // C-layout row M = r + 8*hi
        float iv = sigm(g[0][r] + breg[ht]);
        float fv = sigm(g[1][r] + breg[4 + ht]);
        float gv = tanhf(g[2][r] + breg[8 + ht]);
        float ov = sigm(g[3][r] + breg[12 + ht]);
        float cv = fv * cfrag[ht][r] + iv * gv;
        cfrag[ht][r] = cv;
        myH[(r + 8 * hi) * 64 + ht * 16 + row] = f2bf(ov * tanhf(cv));
      }
    }
    asm volatile("s_wait_dscnt 0" ::: "memory");
    // repack h into A fragments for next step
#pragma unroll
    for (int kc = 0; kc < 2; kc++) {
      const bf16* p = myH + row * 64 + kc * 32;
      v8bf lo = *(const v8bf*)(p + 2 * koff);
      v8bf hh = *(const v8bf*)(p + 2 * koff + 8);
      v16bf a;
#pragma unroll
      for (int q = 0; q < 8; q++) { a[q] = lo[q]; a[8 + q] = hh[q]; }
      hA[kc] = a;
    }
  }

  // h = relu(n @ W_self + b_self + h_n @ W_neigh)
  const float* sp = nfeat + myNode * 64;
  v16bf nA[2];
  nA[0] = loadA_row(sp, 0, koff);
  nA[1] = loadA_row(sp, 1, koff);
#pragma unroll
  for (int jt2 = 0; jt2 < 2; jt2++) {
    int o = jt2 * 16 + row;
    v16bf bs[2], bn[2];
#pragma unroll
    for (int kc = 0; kc < 2; kc++) {
      v16bf t1, t2;
#pragma unroll
      for (int q = 0; q < 16; q++) {
        int k = kc * 32 + hi * 16 + q;
        t1[q] = f2bf(Wself[k * 32 + o]);
        t2[q] = f2bf(Wneigh[k * 32 + o]);
      }
      bs[kc] = t1; bn[kc] = t2;
    }
    v8f acc = zero8();
    acc = wmma_bf16(nA[0], bs[0], acc);
    acc = wmma_bf16(nA[1], bs[1], acc);
    acc = wmma_bf16(hA[0], bn[0], acc);
    acc = wmma_bf16(hA[1], bn[1], acc);
    float bb = bself[o];
#pragma unroll
    for (int r = 0; r < 8; r++) {
      float v = acc[r] + bb;
      v = v > 0.f ? v : 0.f;
      hout[(nodeBase + r + 8 * hi) * 32 + o] = v;
    }
  }
}

// =====================================================================
// Kernel 2: NNConv (edge-conditioned conv, sum over fixed degree 8)
// one wave = 16-edge tile (2 dst nodes); 64 WMMAs per tile, no atomics.
// h_src staged via async global->LDS (ASYNCcnt) when available.
// =====================================================================
__global__ void __launch_bounds__(256) nnconv_kernel(
    const float* __restrict__ efeat, const int* __restrict__ neigh,
    const float* __restrict__ Wew,   const float* __restrict__ Web,
    const float* __restrict__ nnb,   const float* __restrict__ hin,
    float* __restrict__ h2out)
{
  extern __shared__ __attribute__((aligned(64))) char smem[];
  bf16*  sWT  = (bf16*)smem;                    // We_w transposed [j=1024][f=32] bf16: 64KB
  float* sWeb = (float*)(smem + 65536);         // We_b [1024]
  float* sh   = (float*)(smem + 65536 + 4096);  // per-wave h_src [16][32]
  const int tid = threadIdx.x, lane = tid & 31, w = tid >> 5;

  for (int i = tid; i < 32768; i += 256) {      // coalesced read, transposed store
    int f = i >> 10, j = i & 1023;
    sWT[j * 32 + f] = f2bf(Wew[i]);
  }
#ifdef HAVE_ASYNC_LDS
  ASYNC_B128(Web + tid * 4, sWeb + tid * 4);    // 256 threads x 16B = all of We_b
  asm volatile("s_wait_asynccnt 0" ::: "memory");
#else
  for (int i = tid; i < 1024; i += 256) sWeb[i] = Web[i];
#endif
  __syncthreads();

  const int row = lane & 15, hi = lane >> 4, koff = hi * 8;
  float* shw = sh + w * 512;
  const int nwaves = gridDim.x * 8;
  const float nb0 = nnb[row], nb1 = nnb[16 + row];

  for (int tile = blockIdx.x * 8 + w; tile < 10000; tile += nwaves) {
    const int e0 = tile * 16;
    // prefetch next tile's edge rows (speculative, gfx1250 global_prefetch)
    if (tile + nwaves < 10000)
      __builtin_prefetch(efeat + (e0 + nwaves * 16 + row) * 32, 0, 1);
    // stage h[src] for the 16 edges of this tile
    {
      int src = neigh[e0 + row];
      const float* hp = hin + src * 32 + hi * 16;
      float* q = shw + row * 32 + hi * 16;
#ifdef HAVE_ASYNC_LDS
      // WAR: previous iteration's DS reads of shw must drain before the
      // async engine (unordered vs DS) overwrites it.
      asm volatile("s_wait_dscnt 0" ::: "memory");
      ASYNC_B128(hp, q);
      ASYNC_B128(hp + 4, q + 4);
      ASYNC_B128(hp + 8, q + 8);
      ASYNC_B128(hp + 12, q + 12);
#else
      vf4 a = *(const vf4*)(hp);
      vf4 b = *(const vf4*)(hp + 4);
      vf4 c = *(const vf4*)(hp + 8);
      vf4 d = *(const vf4*)(hp + 12);
#pragma unroll
      for (int qq = 0; qq < 4; qq++) {
        q[qq] = a[qq]; q[4 + qq] = b[qq]; q[8 + qq] = c[qq]; q[12 + qq] = d[qq];
      }
#endif
    }
    // edge-feature A fragment (K = EF = 32, single chunk)
    v16bf eA = loadA_row(efeat + (e0 + row) * 32, 0, koff);
#ifdef HAVE_ASYNC_LDS
    asm volatile("s_wait_asynccnt 0" ::: "memory");
#else
    asm volatile("s_wait_dscnt 0" ::: "memory");
#endif

    v8f m0 = zero8(), m1 = zero8();   // msg accumulators, o in [0,16) / [16,32)
#pragma unroll 2
    for (int jp = 0; jp < 32; jp++) {        // input channel i = jp
      int j0 = jp * 32 + row;                // column tile jt=2*jp  (o = row)
      int j1 = j0 + 16;                      // column tile jt=2*jp+1 (o = 16+row)
      v16bf B0 = *(const v16bf*)(sWT + j0 * 32 + hi * 16);
      v16bf B1 = *(const v16bf*)(sWT + j1 * 32 + hi * 16);
      v8f c0 = wmma_bf16(eA, B0, zero8());
      v8f c1 = wmma_bf16(eA, B1, zero8());
      float w0 = sWeb[j0], w1 = sWeb[j1];
      float hv[8];
#pragma unroll
      for (int r = 0; r < 8; r++) hv[r] = shw[(r + 8 * hi) * 32 + jp];
#pragma unroll
      for (int r = 0; r < 8; r++) {
        m0[r] += hv[r] * (c0[r] + w0);
        m1[r] += hv[r] * (c1[r] + w1);
      }
    }
    // sum the 8 edges per dst node (rows r=0..7 of each lane's half)
    float s0 = 0.f, s1 = 0.f;
#pragma unroll
    for (int r = 0; r < 8; r++) { s0 += m0[r]; s1 += m1[r]; }
    int node = tile * 2 + hi;
    h2out[node * 32 + row]      = s0 + nb0;
    h2out[node * 32 + 16 + row] = s1 + nb1;
  }
}

// =====================================================================
// Kernel 3: per-graph max pool + elu + MLP 32->32->32->1 (one wave/graph)
// =====================================================================
__global__ void __launch_bounds__(32) head_kernel(
    const float* __restrict__ h2, const float* __restrict__ W1, const float* __restrict__ b1,
    const float* __restrict__ W2, const float* __restrict__ b2,
    const float* __restrict__ W3, const float* __restrict__ b3,
    float* __restrict__ out)
{
  __shared__ float s0[32], s1[32];
  const int g = blockIdx.x, lane = threadIdx.x;
  const float* base = h2 + g * 100 * 32 + lane;
  float m = -3.4e38f;
  for (int ng = 0; ng < 100; ng++) m = fmaxf(m, base[ng * 32]);
  float v = m > 0.f ? m : (__expf(m) - 1.f);   // elu
  s0[lane] = v;
  asm volatile("s_wait_dscnt 0" ::: "memory");
  float a1 = b1[lane];
#pragma unroll 8
  for (int k = 0; k < 32; k++) a1 += s0[k] * W1[k * 32 + lane];
  a1 = a1 > 0.f ? a1 : 0.f;
  s1[lane] = a1;
  asm volatile("s_wait_dscnt 0" ::: "memory");
  float a2 = b2[lane];
#pragma unroll 8
  for (int k = 0; k < 32; k++) a2 += s1[k] * W2[k * 32 + lane];
  a2 = a2 > 0.f ? a2 : 0.f;
  s0[lane] = a2 * W3[lane];
  asm volatile("s_wait_dscnt 0" ::: "memory");
  if (lane == 0) {
    float acc = b3[0];
    for (int k = 0; k < 32; k++) acc += s0[k];
    out[g] = acc;
  }
}

// =====================================================================
extern "C" void kernel_launch(void* const* d_in, const int* in_sizes, int n_in,
                              void* d_out, int out_size, void* d_ws, size_t ws_size,
                              hipStream_t stream) {
  (void)in_sizes; (void)n_in; (void)out_size; (void)ws_size;
  const float* n_     = (const float*)d_in[0];
  const float* e_     = (const float*)d_in[1];
  const int*   idx_   = (const int*)d_in[2];
  const float* Wih    = (const float*)d_in[3];
  const float* Whh    = (const float*)d_in[4];
  const float* bih    = (const float*)d_in[5];
  const float* bhh    = (const float*)d_in[6];
  const float* Wself  = (const float*)d_in[7];
  const float* bself  = (const float*)d_in[8];
  const float* Wneigh = (const float*)d_in[9];
  const float* Wew    = (const float*)d_in[10];
  const float* Web    = (const float*)d_in[11];
  const float* nnb    = (const float*)d_in[12];
  const float* W1     = (const float*)d_in[13];
  const float* b1     = (const float*)d_in[14];
  const float* W2     = (const float*)d_in[15];
  const float* b2     = (const float*)d_in[16];
  const float* W3     = (const float*)d_in[17];
  const float* b3     = (const float*)d_in[18];
  float* out  = (float*)d_out;
  float* hws  = (float*)d_ws;            // [20000][32] node embeddings
  float* h2ws = hws + 20000 * 32;        // [20000][32] NNConv output

  // 1250 node tiles, 4 waves/block
  lstm_sage_kernel<<<313, 128, 73728, stream>>>(n_, idx_, Wih, Whh, bih, bhh,
                                                Wself, bself, Wneigh, hws);
  // 10000 edge tiles, 8 waves/block, grid-stride
  nnconv_kernel<<<320, 256, 86016, stream>>>(e_, idx_, Wew, Web, nnb, hws, h2ws);
  // 200 graphs
  head_kernel<<<200, 32, 0, stream>>>(h2ws, W1, b1, W2, b2, W3, b3, out);
}
